// Self_Additive_Attention_55860344652422
// MI455X (gfx1250) — compile-verified, matrix operand order checked by reading, not run on hardware
//
#include <hip/hip_runtime.h>
#include <hip/hip_bf16.h>
#include <stdint.h>

// Self-additive attention for MI455X (gfx1250, wave32, WMMA).
// Shapes: BS=64, SL=2048, D=128 for all feature dims.

#define BS 64
#define SL 2048
#define D  128

typedef __attribute__((ext_vector_type(16))) __bf16 v16bf;
typedef __attribute__((ext_vector_type(8)))  __bf16 v8bf;
typedef __attribute__((ext_vector_type(8)))  float  v8f;

// Build a 16-element bf16 A-fragment from a global f32 row.
// CDNA5 16-bit A layout (16x32 MxK): lane<16 -> row r, K = kA+0..7 then kB+0..7
// where (kA,kB) = (k0, k0+16) for lane<16 and (k0+8, k0+24) for lane>=16.
__device__ inline v16bf load_a_frag_f32(const float* __restrict__ row, int kA, int kB) {
    v16bf a;
#pragma unroll
    for (int i = 0; i < 8; ++i) {
        a[i]     = (__bf16)row[kA + i];
        a[8 + i] = (__bf16)row[kB + i];
    }
    return a;
}

// B-fragment (K x 16 tile of W) stored as W^T in LDS: ldsT[n*D + k], bf16.
// Same per-lane K split as the A layout with M<->N.
__device__ inline v16bf load_b_frag_lds(const __bf16* __restrict__ colT, int kA, int kB) {
    v8bf lo = *(const v8bf*)(colT + kA);   // 16B aligned: kA multiple of 8
    v8bf hi = *(const v8bf*)(colT + kB);
    return __builtin_shufflevector(lo, hi, 0,1,2,3,4,5,6,7,8,9,10,11,12,13,14,15);
}

// ---------------------------------------------------------------------------
// K1: scores[b,s] = vq . tanh(query[b,s,:]@Wq + value[b,s,:]@Wk)
// grid = (BS, SL/128), block = 256 (8 waves x 16 rows = 128 seq rows / block)
// ---------------------------------------------------------------------------
__global__ __launch_bounds__(256)
void saa_score_kernel(const float* __restrict__ query,
                      const float* __restrict__ value,
                      const float* __restrict__ Wq,
                      const float* __restrict__ Wk,
                      const float* __restrict__ vq,
                      float* __restrict__ scores)
{
    __shared__ __bf16 ldsWqT[D * D];   // WqT[n*D + k] = Wq[k*D + n]
    __shared__ __bf16 ldsWkT[D * D];
    __shared__ float  ldsVq[D];

    const int b    = blockIdx.x;
    const int tid  = threadIdx.x;
    const int rows0 = blockIdx.y * 128;

    for (int i = tid; i < D * D; i += 256) {
        const int n = i >> 7, k = i & (D - 1);
        ldsWqT[i] = (__bf16)Wq[k * D + n];
        ldsWkT[i] = (__bf16)Wk[k * D + n];
    }
    if (tid < D) ldsVq[tid] = vq[tid];
    __syncthreads();

    const int wave = tid >> 5;
    const int lane = tid & 31;
    const int half = lane >> 4;     // which 16-lane group
    const int r    = lane & 15;

    const int s0 = rows0 + wave * 16;
    const float* qrow = query + ((size_t)b * SL + s0 + r) * D;
    const float* vrow = value + ((size_t)b * SL + s0 + r) * D;

    v8f acc[8] = {};                // 8 N-tiles of a 16x128 C, f32 accum

#pragma unroll
    for (int ks = 0; ks < 4; ++ks) {            // K = 128 in steps of 32
        const int kA = ks * 32 + half * 8;
        const int kB = ks * 32 + 16 + half * 8;
        const v16bf aq = load_a_frag_f32(qrow, kA, kB);
        const v16bf av = load_a_frag_f32(vrow, kA, kB);
#pragma unroll
        for (int nt = 0; nt < 8; ++nt) {
            const __bf16* colQ = ldsWqT + (nt * 16 + r) * D;
            const __bf16* colK = ldsWkT + (nt * 16 + r) * D;
            const v16bf bq = load_b_frag_lds(colQ, kA, kB);
            const v16bf bk = load_b_frag_lds(colK, kA, kB);
            acc[nt] = __builtin_amdgcn_wmma_f32_16x16x32_bf16(
                false, aq, false, bq, (short)0, acc[nt], false, false);
            acc[nt] = __builtin_amdgcn_wmma_f32_16x16x32_bf16(
                false, av, false, bk, (short)0, acc[nt], false, false);
        }
    }

    // score[m] = sum_n tanh(C[m,n]) * vq[n].
    // C layout: VGPR i, lane<16 -> (M=i, N=lane); lane>=16 -> (M=8+i, N=lane-16).
    float part[8];
#pragma unroll
    for (int i = 0; i < 8; ++i) part[i] = 0.0f;
#pragma unroll
    for (int nt = 0; nt < 8; ++nt) {
        const float w = ldsVq[nt * 16 + r];
#pragma unroll
        for (int i = 0; i < 8; ++i) part[i] += tanhf(acc[nt][i]) * w;
    }
    // reduce across the 16 lanes of each half-group (wave32 shuffles)
#pragma unroll
    for (int m = 8; m >= 1; m >>= 1) {
#pragma unroll
        for (int i = 0; i < 8; ++i) part[i] += __shfl_xor(part[i], m, 32);
    }
    // lanes r<8 of each half write rows (half*8 + r)
    if (r < 8)
        scores[(size_t)b * SL + s0 + half * 8 + r] = part[r];
}

// ---------------------------------------------------------------------------
// K2: masked softmax over SL per batch (in-place in workspace); zero ctx.
// grid = BS, block = 256
// ---------------------------------------------------------------------------
__global__ __launch_bounds__(256)
void saa_softmax_kernel(const float* __restrict__ scores_in,
                        const uint8_t* __restrict__ mask,   // jnp bool: 1 byte
                        float* __restrict__ weights,        // may alias scores_in
                        float* __restrict__ ctx)
{
    __shared__ float sdata[SL];
    __shared__ float red[256];
    const int b = blockIdx.x, tid = threadIdx.x;

    float lmax = -3.4e38f;
    for (int s = tid; s < SL; s += 256) {
        const float v = mask[(size_t)b * SL + s] ? -1.0e6f
                                                 : scores_in[(size_t)b * SL + s];
        sdata[s] = v;
        lmax = fmaxf(lmax, v);
    }
    red[tid] = lmax; __syncthreads();
    for (int o = 128; o > 0; o >>= 1) {
        if (tid < o) red[tid] = fmaxf(red[tid], red[tid + o]);
        __syncthreads();
    }
    const float mx = red[0]; __syncthreads();

    float lsum = 0.0f;
    for (int s = tid; s < SL; s += 256) {
        const float e = __expf(sdata[s] - mx);
        sdata[s] = e;
        lsum += e;
    }
    red[tid] = lsum; __syncthreads();
    for (int o = 128; o > 0; o >>= 1) {
        if (tid < o) red[tid] += red[tid + o];
        __syncthreads();
    }
    const float inv = 1.0f / red[0];
    for (int s = tid; s < SL; s += 256)
        weights[(size_t)b * SL + s] = sdata[s] * inv;
    if (tid < D) ctx[b * D + tid] = 0.0f;    // zero accumulator for K3
}

// ---------------------------------------------------------------------------
// K3: ctx[b,v] = sum_s weight[b,s] * value[b,s,v]
// grid = (BS, SL/256), block = 256 (2 groups x 128 columns)
// ---------------------------------------------------------------------------
__global__ __launch_bounds__(256)
void saa_ctx_kernel(const float* __restrict__ value,
                    const float* __restrict__ weights,
                    float* __restrict__ ctx)
{
    __shared__ float wsh[256];
    __shared__ float partial[256];
    const int b = blockIdx.x, chunk = blockIdx.y;
    const int tid = threadIdx.x;
    const int v = tid & (D - 1);
    const int g = tid >> 7;          // 0 or 1

    wsh[tid] = weights[(size_t)b * SL + chunk * 256 + tid];
    __syncthreads();

    const float* base = value + ((size_t)b * SL + chunk * 256) * D;
    float acc = 0.0f;
    for (int s = g; s < 256; s += 2)
        acc += wsh[s] * base[(size_t)s * D + v];

    partial[tid] = acc; __syncthreads();
    if (g == 0)
        atomicAdd(&ctx[b * D + v], acc + partial[128 + v]);
}

// ---------------------------------------------------------------------------
// K4: res[b,o] = sum_v ctx[b,v] * Wv[v,o]   (tiny 64x128x128 matvec)
// grid = BS, block = 128
// ---------------------------------------------------------------------------
__global__ __launch_bounds__(128)
void saa_out_kernel(const float* __restrict__ ctx,
                    const float* __restrict__ Wv,
                    float* __restrict__ out)
{
    __shared__ float c[D];
    const int b = blockIdx.x, o = threadIdx.x;
    c[o] = ctx[b * D + o];
    __syncthreads();
    float acc = 0.0f;
#pragma unroll 4
    for (int v = 0; v < D; ++v) acc += c[v] * Wv[v * D + o];
    out[b * D + o] = acc;
}

// ---------------------------------------------------------------------------
extern "C" void kernel_launch(void* const* d_in, const int* in_sizes, int n_in,
                              void* d_out, int out_size, void* d_ws, size_t ws_size,
                              hipStream_t stream)
{
    // setup_inputs() order: value, query, Wq, Wk, Wv, vq, mask
    const float*   value = (const float*)d_in[0];
    const float*   query = (const float*)d_in[1];
    const float*   Wq    = (const float*)d_in[2];
    const float*   Wk    = (const float*)d_in[3];
    const float*   Wv    = (const float*)d_in[4];
    const float*   vq    = (const float*)d_in[5];
    const uint8_t* mask  = (const uint8_t*)d_in[6];
    float*         out   = (float*)d_out;

    float* scores = (float*)d_ws;          // BS*SL floats (reused as weights)
    float* ctx    = scores + (size_t)BS * SL;  // BS*D floats

    saa_score_kernel<<<dim3(BS, SL / 128), 256, 0, stream>>>(
        query, value, Wq, Wk, vq, scores);
    saa_softmax_kernel<<<BS, 256, 0, stream>>>(scores, mask, scores, ctx);
    saa_ctx_kernel<<<dim3(BS, SL / 256), 256, 0, stream>>>(value, scores, ctx);
    saa_out_kernel<<<BS, 128, 0, stream>>>(ctx, Wv, out);
}
